// Scatter2Operator_5531917878013
// MI455X (gfx1250) — compile-verified, matrix operand order checked by reading, not run on hardware
//
#include <hip/hip_runtime.h>
#include <hip/hip_bf16.h>
#include <math.h>

// ---------------------------------------------------------------------------
// Scatter2Operator for MI455X (gfx1250, wave32, WMMA).
//
// Roofline: ~147 GFLOP vs ~0.4 GB HBM traffic -> matrix-core bound.
// All 4 GEMMs run through v_wmma_f32_16x16x32_f16 (f32 accum).
// Fused per-64-row-tile kernel keeps x2 tile + pred + scores in 304KB of LDS
// (CDNA5 WGP has 320KB); f16 weights (~7MB) stay resident in the 192MB L2.
// Head-PAIR processing: every wave owns 4 independent WMMA accumulator chains
// sharing one B fragment per K-step (max ILP, no duplicate B loads).
// ---------------------------------------------------------------------------

typedef __attribute__((ext_vector_type(16))) _Float16 v16h;
typedef __attribute__((ext_vector_type(8)))  float    v8f;

#define B_DIM  8
#define N_DIM  4096
#define G_DIM  64
#define H_DIM  16
#define DH_DIM 64
#define D_DIM  1024
#define TILE_N 64

// ws layout (bytes) -- total 8 MB
#define WK_OFF 0u                  // Wk f16  [1024][1024]  2 MB
#define WP_OFF (2u*1024u*1024u)    // Wp f16  [1024][1024]  2 MB
#define WO_OFF (4u*1024u*1024u)    // Wo f16  [1024][1024]  2 MB
#define X0_OFF (6u*1024u*1024u)    // x0 f16  [8][64][1024] 1 MB
#define PT_OFF (7u*1024u*1024u)    // pT f16  [8][16][64][64] 1 MB

// ISA 16-bit A/B fragment (16x32): lane l holds row (l&15); K chunks
// {kBase + (l>>4)*8 .. +7} and {kBase + 16 + (l>>4)*8 .. +7}, 16B each.
__device__ __forceinline__ v16h load_frag(const _Float16* base, int strideElems,
                                          int lane, int kBase) {
  const int r  = lane & 15;
  const int hi = lane >> 4;
  const _Float16* p0 = base + r * strideElems + kBase + hi * 8;
  union { v16h v; uint4 q[2]; } f;
  f.q[0] = *reinterpret_cast<const uint4*>(p0);
  f.q[1] = *reinterpret_cast<const uint4*>(p0 + 16);
  return f.v;
}

__device__ __forceinline__ v8f wmma16(v16h a, v16h b, v8f c) {
  // D = A(16x32) * B(32x16) + C, f32 accumulate
  return __builtin_amdgcn_wmma_f32_16x16x32_f16(false, a, false, b,
                                                (short)0, c, false, false);
}

// ---------------------------------------------------------------------------
// Kernel 0: f32 -> f16 conversion (weights + x0), grid-stride
// ---------------------------------------------------------------------------
__global__ void cvt_f32_f16(const float* __restrict__ src,
                            _Float16* __restrict__ dst, int n) {
  int i = blockIdx.x * blockDim.x + threadIdx.x;
  const int stride = gridDim.x * blockDim.x;
  for (; i < n; i += stride) dst[i] = (_Float16)src[i];
}

// ---------------------------------------------------------------------------
// Kernel 1: p = x0 @ Wp^T, stored transposed per head: pT[b][h][dh][g] (f16)
// grid.x = B*H; 8 waves; wave w -> colTile (w&3), rowTiles {2*(w>>2), +1}
// ---------------------------------------------------------------------------
__global__ void premise_kernel(const _Float16* __restrict__ x0h,
                               const _Float16* __restrict__ wp,
                               _Float16* __restrict__ pT) {
  const int b = blockIdx.x >> 4;
  const int h = blockIdx.x & 15;
  const int tid = threadIdx.x;
  const int w = tid >> 5, lane = tid & 31;
  const int tc = w & 3;               // dh col tile (0..3)
  const int tr0 = (w >> 2) * 2;       // g row tiles tr0, tr0+1

  const _Float16* aBase0 = x0h + (b * G_DIM + tr0 * 16) * D_DIM;
  const _Float16* aBase1 = aBase0 + 16 * D_DIM;
  const _Float16* bBase  = wp + (h * DH_DIM + tc * 16) * D_DIM;

  v8f acc0 = {}, acc1 = {};
  for (int kb = 0; kb < D_DIM; kb += 32) {
    v16h bf = load_frag(bBase, D_DIM, lane, kb);
    v16h a0 = load_frag(aBase0, D_DIM, lane, kb);
    v16h a1 = load_frag(aBase1, D_DIM, lane, kb);
    acc0 = wmma16(a0, bf, acc0);
    acc1 = wmma16(a1, bf, acc1);
  }

  // D layout: lane -> col N=(lane&15); vgpr v -> row M = v + 8*(lane>>4)
  const int c = lane & 15, hi = lane >> 4;
  const int dh = tc * 16 + c;
  _Float16* outRow = pT + ((b * H_DIM + h) * DH_DIM + dh) * G_DIM;
#pragma unroll
  for (int v = 0; v < 8; ++v) {
    outRow[tr0 * 16 + v + 8 * hi]        = (_Float16)acc0[v];
    outRow[(tr0 + 1) * 16 + v + 8 * hi]  = (_Float16)acc1[v];
  }
}

// ---------------------------------------------------------------------------
// Kernel 2: fused scores -> softmax -> gather -> output GEMM per 64-row tile.
// grid = (N/64, B); 256 threads (8 wave32). Heads processed in PAIRS.
// Dynamic LDS: xt f16 64x1024 | pred f16 64x1024 | sbuf f32 64x128 | kbuf f16 64x128
// ---------------------------------------------------------------------------
__global__ void fused_kernel(const float* __restrict__ x2,
                             const _Float16* __restrict__ wk,
                             const _Float16* __restrict__ wo,
                             const _Float16* __restrict__ pT,
                             const float* __restrict__ bo,
                             float* __restrict__ out) {
  extern __shared__ char smem_raw[];
  _Float16* xt    = (_Float16*)(smem_raw);                    // 131072 B
  _Float16* predL = (_Float16*)(smem_raw + 131072);           // 131072 B
  float*    sbuf  = (float*)   (smem_raw + 262144);           // 64x128 f32: 32768 B
  _Float16* kbuf  = (_Float16*)(smem_raw + 262144 + 32768);   // 64x128 f16: 16384 B

  const int b     = blockIdx.y;
  const int nBase = blockIdx.x * TILE_N;
  const int tid   = threadIdx.x;
  const int w     = tid >> 5, lane = tid & 31;
  const int hl    = w >> 2;           // head-local index within pair (0/1)
  const int tc    = w & 3;            // col tile within that head (0..3)

  // --- load x2 tile [64 x 1024] f32 -> f16 LDS (coalesced float4) ---
  {
    const float4* src = (const float4*)(x2 + ((size_t)(b * N_DIM + nBase)) * D_DIM);
    for (int i = tid; i < TILE_N * D_DIM / 4; i += blockDim.x) {
      float4 f = src[i];
      _Float16* d = xt + i * 4;
      d[0] = (_Float16)f.x; d[1] = (_Float16)f.y;
      d[2] = (_Float16)f.z; d[3] = (_Float16)f.w;
    }
  }
  __syncthreads();

  const int c  = lane & 15;
  const int hi = lane >> 4;

  // ------------- per head-pair: scores -> softmax -> gather -------------
  for (int hp = 0; hp < H_DIM / 2; ++hp) {
    const int h = hp * 2 + hl;        // this wave's head within the pair

    // Stage 1: S = Xtile @ Wk_{h}^T. Wave owns col tile (h, tc) x 4 row tiles:
    // one shared B frag + 4 independent WMMA chains per K-step.
    {
      v8f acc[4] = {{}, {}, {}, {}};
      const _Float16* bBase = wk + (h * G_DIM + tc * 16) * D_DIM;
      for (int kb = 0; kb < D_DIM; kb += 32) {
        v16h bf = load_frag(bBase, D_DIM, lane, kb);
#pragma unroll
        for (int tr = 0; tr < 4; ++tr) {
          v16h a = load_frag(xt + (tr * 16) * D_DIM, D_DIM, lane, kb);
          acc[tr] = wmma16(a, bf, acc[tr]);
        }
      }
      const int col = w * 16 + c;     // 0..127 across the head pair
#pragma unroll
      for (int tr = 0; tr < 4; ++tr)
#pragma unroll
        for (int v = 0; v < 8; ++v)
          sbuf[(tr * 16 + v + 8 * hi) * 128 + col] = acc[tr][v];
    }
    __syncthreads();

    // Stage 2: row softmax over G=64 for both heads -> kbuf (f16).
    // 128 active threads (waves 0-3, wave-uniform predicate).
    if (tid < 2 * TILE_N) {
      const int row = tid & 63;
      const int hc  = (tid >> 6) * 64;  // which head's 64-col slab
      float* srow = sbuf + row * 128 + hc;
      float m = -3.4e38f;
      for (int g = 0; g < G_DIM; ++g) m = fmaxf(m, srow[g]);
      float s = 0.f;
      for (int g = 0; g < G_DIM; ++g) {
        float e = __expf(srow[g] - m);
        srow[g] = e;
        s += e;
      }
      const float inv = 1.0f / s;
      _Float16* krow = kbuf + row * 128 + hc;
      for (int g = 0; g < G_DIM; ++g) krow[g] = (_Float16)(srow[g] * inv);
    }
    __syncthreads();

    // Stage 3: pred_h = K_h(64x64) @ P_h(64x64); B from pT (K=g contiguous).
    // Same wave->tile map: 4 row tiles per wave, one shared B frag.
    {
      v8f acc[4] = {{}, {}, {}, {}};
      const _Float16* bBase = pT + ((b * H_DIM + h) * DH_DIM + tc * 16) * G_DIM;
#pragma unroll
      for (int kb = 0; kb < G_DIM; kb += 32) {
        v16h bf = load_frag(bBase, G_DIM, lane, kb);
#pragma unroll
        for (int tr = 0; tr < 4; ++tr) {
          v16h a = load_frag(kbuf + (tr * 16) * 128 + hl * 64, 128, lane, kb);
          acc[tr] = wmma16(a, bf, acc[tr]);
        }
      }
      const int col = h * DH_DIM + tc * 16 + c;
#pragma unroll
      for (int tr = 0; tr < 4; ++tr)
#pragma unroll
        for (int v = 0; v < 8; ++v)
          predL[(tr * 16 + v + 8 * hi) * D_DIM + col] = (_Float16)acc[tr][v];
    }
    // No barrier needed here: next pair's post-stage-1 barrier orders
    // sbuf/kbuf reuse (every wave finishes stage 3 before reaching it).
  }
  __syncthreads();

  // ---------------- Stage 4: out = pred @ Wo^T + bo ----------------
  // wave w owns col tiles {8w .. 8w+7}; 4 row tiles each; B frag reused x4.
  for (int tcl = 0; tcl < 8; ++tcl) {
    const int ct = w * 8 + tcl;               // output col tile (0..63)
    v8f acc[4] = {{}, {}, {}, {}};
    const _Float16* bBase = wo + (ct * 16) * D_DIM;
    for (int kb = 0; kb < D_DIM; kb += 32) {
      v16h bf = load_frag(bBase, D_DIM, lane, kb);
#pragma unroll
      for (int tr = 0; tr < 4; ++tr) {
        v16h a = load_frag(predL + (tr * 16) * D_DIM, D_DIM, lane, kb);
        acc[tr] = wmma16(a, bf, acc[tr]);
      }
    }
    const int d = ct * 16 + c;
    const float bias = bo[d];
#pragma unroll
    for (int tr = 0; tr < 4; ++tr) {
#pragma unroll
      for (int v = 0; v < 8; ++v) {
        const int row = tr * 16 + v + 8 * hi;
        out[((size_t)(b * N_DIM + nBase + row)) * D_DIM + d] = acc[tr][v] + bias;
      }
    }
  }
}

// ---------------------------------------------------------------------------
extern "C" void kernel_launch(void* const* d_in, const int* in_sizes, int n_in,
                              void* d_out, int out_size, void* d_ws, size_t ws_size,
                              hipStream_t stream) {
  const float* x0 = (const float*)d_in[0];   // [8, 64, 1024]
  const float* x2 = (const float*)d_in[1];   // [8, 4096, 1024]
  const float* Wk = (const float*)d_in[2];   // [1024, 1024]
  const float* Wp = (const float*)d_in[3];   // [1024, 1024]
  const float* Wo = (const float*)d_in[4];   // [1024, 1024]
  const float* bo = (const float*)d_in[5];   // [1024]
  float* out = (float*)d_out;

  char* ws = (char*)d_ws;
  _Float16* wk16 = (_Float16*)(ws + WK_OFF);
  _Float16* wp16 = (_Float16*)(ws + WP_OFF);
  _Float16* wo16 = (_Float16*)(ws + WO_OFF);
  _Float16* x016 = (_Float16*)(ws + X0_OFF);
  _Float16* pT   = (_Float16*)(ws + PT_OFF);

  // Stage A: downconvert weights + x0 to f16 (~7 MB total -> L2 resident)
  cvt_f32_f16<<<256, 256, 0, stream>>>(Wk, wk16, D_DIM * D_DIM);
  cvt_f32_f16<<<256, 256, 0, stream>>>(Wp, wp16, D_DIM * D_DIM);
  cvt_f32_f16<<<256, 256, 0, stream>>>(Wo, wo16, D_DIM * D_DIM);
  cvt_f32_f16<<<256, 256, 0, stream>>>(x0, x016, B_DIM * G_DIM * D_DIM);

  // Stage B: premise projection, transposed per head for contiguous-K B frags
  premise_kernel<<<B_DIM * H_DIM, 256, 0, stream>>>(x016, wp16, pT);

  // Stage C: fused scores/softmax/gather/output per 64-row tile
  const size_t lds = 131072 + 131072 + 32768 + 16384;  // 304 KB of 320 KB WGP LDS
  fused_kernel<<<dim3(N_DIM / TILE_N, B_DIM), 256, lds, stream>>>(
      x2, wk16, wo16, pT, bo, out);
}
